// ReflexMemory_87213605912730
// MI455X (gfx1250) — matrix operand order repaired and checked by reading, not compiled, not run on hardware
//
#include <hip/hip_runtime.h>
#include <hip/hip_bf16.h>

// ReflexMemory lookup on gfx1250.
// Memory-bound on stored_hashes (410 MB @ 23.3 TB/s ~ 17.6 us floor).
// The Hamming matmul (16x100000x1024, 8 FLOP/byte -> ~186 TFLOP/s needed)
// runs on v_wmma_f32_16x16x32_f16. Trick: A is stored as +/-1 (exact in f16),
// so the WMMA accumulator directly yields 2*dots - ssum and the per-row
// stored-hash sums need no VALU work at all:
//   sim = 1 - (popcount(ph) - c) / H
// Streaming loads of stored_hashes are non-temporal (NT hint) so the 410 MB
// stream does not evict the reused ph fragments from the 192 MB L2.

typedef __attribute__((ext_vector_type(16))) _Float16 v16h;
typedef __attribute__((ext_vector_type(8)))  _Float16 v8h;
typedef __attribute__((ext_vector_type(8)))  float    v8f;
typedef __attribute__((ext_vector_type(4)))  float    v4f;

#define B_DIM 16
#define D_DIM 512
#define H_DIM 1024
#define N_DIM 100000
#define WAVES 8
#define TILES_TOTAL ((N_DIM + 15) / 16)   // 6250

// Workspace layout:
//   [0,     32768) : ph as f16 in {-1,+1}, row-major 16 x 1024
//   [32768, 32832) : phsum (count of +1 bits), 16 f32
//   [32832, 32960) : global argmax keys, 16 u64

// ---------------------------------------------------------------------------
// Kernel A: ph = sign(pattern @ proj) as f16 {-1,+1}; phsum[b] = #(+1) bits.
// 1024 threads total (4 blocks x 256), one h column per thread, 16 batch accs.
// ---------------------------------------------------------------------------
__global__ __launch_bounds__(256) void hash_kernel(
    const float* __restrict__ pattern, const float* __restrict__ proj,
    _Float16* __restrict__ ph, float* __restrict__ phsum) {
  __shared__ float pat[B_DIM * D_DIM];   // 32 KB of 320 KB LDS
  __shared__ float psum[B_DIM];
  const int tid = threadIdx.x;
  for (int i = tid; i < B_DIM * D_DIM; i += 256) pat[i] = pattern[i];
  if (tid < B_DIM) psum[tid] = 0.0f;
  __syncthreads();

  const int h = blockIdx.x * 256 + tid;
  float acc[B_DIM];
#pragma unroll
  for (int b = 0; b < B_DIM; ++b) acc[b] = 0.0f;
  for (int d = 0; d < D_DIM; ++d) {
    const float p = proj[(size_t)d * H_DIM + h];   // coalesced across threads
#pragma unroll
    for (int b = 0; b < B_DIM; ++b) acc[b] = fmaf(pat[b * D_DIM + d], p, acc[b]);
  }
#pragma unroll
  for (int b = 0; b < B_DIM; ++b) {
    const bool on = acc[b] > 0.0f;
    ph[b * H_DIM + h] = on ? (_Float16)1.0f : (_Float16)-1.0f;
    if (on) atomicAdd(&psum[b], 1.0f);             // ds_add_f32
  }
  __syncthreads();
  if (tid < B_DIM) atomicAdd(&phsum[tid], psum[tid]);
}

// ---------------------------------------------------------------------------
// Kernel B: per-wave 16-row tile of stored_hashes; WMMA over H with +/-1 A,
// sim + packed-key argmax. Block-level LDS reduction -> 16 atomicMax.
// ---------------------------------------------------------------------------
__global__ __launch_bounds__(256) void sim_kernel(
    const float* __restrict__ stored, const _Float16* __restrict__ ph,
    const float* __restrict__ phsum, unsigned long long* __restrict__ gkeys) {
  __shared__ unsigned long long lkeys[WAVES * 16];
  const int tid     = threadIdx.x;
  const int wave    = tid >> 5;
  const int lane    = tid & 31;
  const int half_id = lane >> 4;     // 0: lanes 0-15, 1: lanes 16-31
  const int lm      = lane & 15;
  const int tile    = blockIdx.x * WAVES + wave;

  unsigned long long mykeys[8];
#pragma unroll
  for (int r = 0; r < 8; ++r) mykeys[r] = 0ull;

  const int row = tile * 16 + lm;    // this lane's stored-hash row (B col N=lm)

  if (tile < TILES_TOTAL) {
    const int rowc = row < N_DIM ? row : (N_DIM - 1);
    // B fragment source: kb = half_id*8; elems 0..7 = K kb..kb+7, 8..15 = K 16+kb..23+kb.
    const float* srow = stored + (size_t)rowc * H_DIM + half_id * 8;
    // A fragment source (ISA 16-bit 16x32 A layout): per lane M=lm,
    // kb = half_id*8; halves 0..7 = K kb..kb+7, halves 8..15 = K 16+kb..23+kb.
    const _Float16* arow = ph + lm * H_DIM + half_id * 8;

    v8f c = {};
    for (int kk = 0; kk < H_DIM; kk += 32) {
      // --- B: two runs of 8 consecutive f32 of one stored row, streamed non-temporally.
      v4f q0 = __builtin_nontemporal_load((const v4f*)(srow + kk));
      v4f q1 = __builtin_nontemporal_load((const v4f*)(srow + kk + 4));
      v4f q2 = __builtin_nontemporal_load((const v4f*)(srow + kk + 16));
      v4f q3 = __builtin_nontemporal_load((const v4f*)(srow + kk + 20));
      v16h bv;
#pragma unroll
      for (int j = 0; j < 4; ++j) {
        bv[j]      = (_Float16)q0[j];
        bv[4 + j]  = (_Float16)q1[j];
        bv[8 + j]  = (_Float16)q2[j];
        bv[12 + j] = (_Float16)q3[j];
      }
      // --- A: two 16B f16 loads matching the A VGPR layout (L1-resident).
      union { v16h v; struct { v8h lo; v8h hi; } h; } au;
      au.h.lo = *(const v8h*)(arow + kk);
      au.h.hi = *(const v8h*)(arow + kk + 16);
      // D = A x B + C  (16x16x32 f16 -> f32); c accumulates 2*dots - ssum.
      c = __builtin_amdgcn_wmma_f32_16x16x32_f16(
          false, au.v, false, bv, (short)0, c, false, false);
    }

    if (row < N_DIM) {
      const float invH = 1.0f / (float)H_DIM;
#pragma unroll
      for (int r = 0; r < 8; ++r) {
        // C layout: lanes 0-15 -> M=r, lanes 16-31 -> M=8+r; N = lm.
        const int m = half_id * 8 + r;
        float sim = 1.0f - (phsum[m] - c[r]) * invH;
        if (sim < 0.0f) sim = 0.0f;   // keep u64 packing monotonic
        const unsigned int sb = __float_as_uint(sim);
        mykeys[r] = ((unsigned long long)sb << 32) |
                    (unsigned long long)(0xFFFFFFFFu - (unsigned int)row);
      }
    }
  }

  // Butterfly max within each 16-lane half (reduces over the 16 tile rows).
#pragma unroll
  for (int r = 0; r < 8; ++r) {
    unsigned long long k = mykeys[r];
#pragma unroll
    for (int s = 1; s < 16; s <<= 1) {
      unsigned long long o = __shfl_xor(k, s, 32);
      k = o > k ? o : k;
    }
    mykeys[r] = k;
  }
  if (lm == 0) {
#pragma unroll
    for (int r = 0; r < 8; ++r) lkeys[wave * 16 + half_id * 8 + r] = mykeys[r];
  }
  __syncthreads();

  if (tid < 16) {
    unsigned long long best = 0ull;
#pragma unroll
    for (int w = 0; w < WAVES; ++w) {
      unsigned long long k = lkeys[w * 16 + tid];
      best = k > best ? k : best;
    }
    if (best) atomicMax(&gkeys[tid], best);
  }
}

// ---------------------------------------------------------------------------
// Kernel C: decode keys -> best_sim/best_idx, threshold gate, gather preds.
// ---------------------------------------------------------------------------
__global__ __launch_bounds__(512) void finalize_kernel(
    const unsigned long long* __restrict__ gkeys,
    const float* __restrict__ predictions, float* __restrict__ out) {
  __shared__ int   s_idx[16];
  __shared__ float s_gate[16];
  const int tid = threadIdx.x;
  if (tid < 16) {
    const unsigned long long k = gkeys[tid];
    const unsigned int idx = 0xFFFFFFFFu - (unsigned int)(k & 0xFFFFFFFFull);
    const float sim = __uint_as_float((unsigned int)(k >> 32));
    out[B_DIM * D_DIM + tid]      = sim;         // best_sim
    out[B_DIM * D_DIM + 16 + tid] = (float)idx;  // best_idx (as float)
    s_idx[tid]  = (int)idx;
    s_gate[tid] = sim >= 0.9f ? 1.0f : 0.0f;
  }
  __syncthreads();
#pragma unroll
  for (int m = 0; m < B_DIM; ++m)
    out[m * D_DIM + tid] =
        predictions[(size_t)s_idx[m] * D_DIM + tid] * s_gate[m];
}

// ---------------------------------------------------------------------------
extern "C" void kernel_launch(void* const* d_in, const int* in_sizes, int n_in,
                              void* d_out, int out_size, void* d_ws,
                              size_t ws_size, hipStream_t stream) {
  const float* pattern = (const float*)d_in[0];   // (16, 512)
  const float* proj    = (const float*)d_in[1];   // (512, 1024)
  const float* stored  = (const float*)d_in[2];   // (100000, 1024)
  const float* preds   = (const float*)d_in[3];   // (100000, 512)
  float* out = (float*)d_out;

  char* ws = (char*)d_ws;
  _Float16* ph   = (_Float16*)ws;
  float* phsum   = (float*)(ws + 32768);
  unsigned long long* gkeys = (unsigned long long*)(ws + 32832);

  // Zero phsum + argmax keys (graph-capture safe).
  hipMemsetAsync(ws + 32768, 0, 64 + 128, stream);

  hash_kernel<<<H_DIM / 256, 256, 0, stream>>>(pattern, proj, ph, phsum);

  const int blocks = (TILES_TOTAL + WAVES - 1) / WAVES;  // 782
  sim_kernel<<<blocks, 256, 0, stream>>>(stored, ph, phsum, gkeys);

  finalize_kernel<<<1, 512, 0, stream>>>(gkeys, preds, out);
}
